// HONESTER_8658654068988
// MI455X (gfx1250) — compile-verified, matrix operand order checked by reading, not run on hardware
//
#include <hip/hip_runtime.h>
#include <math.h>

// ---------------- problem constants ----------------
#define BB1   1024
#define NB    20          // neighbors (K in the reference)
#define DD    128         // D
#define RR    3           // R
#define DMM   384         // DM = 3*D
#define DKH   192         // DK = DM / NH
#define RB    4           // batch rows per workgroup
#define KROWS (RB*NB)     // 80 k-rows
#define AROWS 96          // 80 k-rows + 16 (padded) q-rows
#define NEGV  (-10000000000.0f)

typedef __attribute__((ext_vector_type(16))) __bf16 v16bf;
typedef __attribute__((ext_vector_type(8)))  __bf16 v8bf;
typedef __attribute__((ext_vector_type(8)))  float  v8f;

// ---------------- LDS layout (bytes) ----------------
#define SM_A     0          // 96*384  bf16 = 73728
#define SM_KH    73728      // 80*384  bf16 = 61440
#define SM_VH    135168     // 80*384  bf16 = 61440
#define SM_QH    196608     // 16*384  f32  = 24576
#define SM_X     221184     // 16*384  f32  = 24576
#define SM_XB    245760     // 16*512  bf16 = 16384
#define SM_ATTN  262144     // 4*2*32  f32  = 1024
#define SMEM_BYTES 263168

// ---- bf16 WMMA fragment loader (A from LDS or B from transposed weights) ----
// ISA 16-bit 16x32 layout: lane&15 selects row (M for A, N for B-transposed),
// half-wave selects K-subblock: elems 0..7 = K[k0+8h .. +7], 8..15 = K[k0+16+8h .. +7]
__device__ __forceinline__ v16bf frag_ld(const __bf16* __restrict__ base, int ld,
                                         int r0, int k0, int lane) {
  const int r  = r0 + (lane & 15);
  const int kk = k0 + ((lane >> 4) << 3);
  const __bf16* p = base + r * ld + kk;
  v8bf lo = *(const v8bf*)(p);
  v8bf hi = *(const v8bf*)(p + 16);
  return __builtin_shufflevector(lo, hi, 0,1,2,3,4,5,6,7,8,9,10,11,12,13,14,15);
}

__device__ __forceinline__ v8f wmma_bf16(v16bf a, v16bf b, v8f c) {
  // D = A(16x32) * B(32x16) + C, f32 accumulate
  return __builtin_amdgcn_wmma_f32_16x16x32_bf16(false, a, false, b, (short)0, c, false, false);
}

// ---------------- weight convert + transpose: f32 (L,K,N) -> bf16 (L,N,K) ----------------
__global__ void cvt_transpose_bf16(const float* __restrict__ src, __bf16* __restrict__ dst,
                                   int L, int Kd, int Nd) {
  long idx = (long)blockIdx.x * blockDim.x + threadIdx.x;
  long total = (long)L * Kd * Nd;
  if (idx >= total) return;
  int per = Kd * Nd;
  int l = (int)(idx / per);
  int rem = (int)(idx % per);
  int k = rem / Nd, n = rem % Nd;
  dst[((long)l * Nd + n) * Kd + k] = (__bf16)src[idx];
}

// ---------------- fused attention layer ----------------
__global__ __launch_bounds__(256)
void tgat_attn_layer(
    int Bb,
    const float* __restrict__ node_table,
    const int*   __restrict__ src_gidx,   // gather idx for src (or null)
    const float* __restrict__ src_dense,  // dense src features (or null)
    const int*   __restrict__ seq_gidx,   // gather idx for neighbors (or null)
    const float* __restrict__ seq_dense,  // dense neighbor features (or null)
    const int*   __restrict__ mask_idx,   // Bb*NB node ids (mask = id==0)
    const float* __restrict__ row_time,   // Bb
    const float* __restrict__ seq_t,      // Bb*NB
    const float* __restrict__ seq_rel,    // Bb*NB*RR
    const float* __restrict__ basis_freq,
    const float* __restrict__ phase,
    const float* __restrict__ rel_W,      // RR*DD
    const float* __restrict__ rel_b,      // DD
    const __bf16* __restrict__ wqT,       // DMMxDMM, N-major
    const __bf16* __restrict__ wkT,
    const __bf16* __restrict__ wvT,
    const __bf16* __restrict__ fcWT,
    const float* __restrict__ fc_b,
    const float* __restrict__ ln_g,
    const float* __restrict__ ln_b,
    const __bf16* __restrict__ mg1T,      // 128x512, N-major
    const float* __restrict__ mg1_b,
    const __bf16* __restrict__ mg2T,      // 128x128, N-major
    const float* __restrict__ mg2_b,
    float* __restrict__ outp)
{
  extern __shared__ char smem[];
  __bf16* sA   = (__bf16*)(smem + SM_A);
  __bf16* sKH  = (__bf16*)(smem + SM_KH);
  __bf16* sVH  = (__bf16*)(smem + SM_VH);
  float*  sQH  = (float*) (smem + SM_QH);
  float*  sX   = (float*) (smem + SM_X);
  __bf16* sXB  = (__bf16*)(smem + SM_XB);
  float*  sATT = (float*) (smem + SM_ATTN);

  const int tid  = threadIdx.x;
  const int lane = tid & 31;
  const int wave = tid >> 5;
  const int row0 = blockIdx.x * RB;
  (void)Bb;

  // ---- Stage 0: zero bf16 GEMM-A staging buffer (16x512) ----
  for (int i = tid; i < (16*512*2)/4; i += 256) ((unsigned*)sXB)[i] = 0u;

  // ---- Stage 1: build stacked [K(80 rows) ; Q(16 rows)] x 384 in bf16 ----
  for (int idx = tid; idx < AROWS*DMM; idx += 256) {
    int r = idx / DMM, c = idx % DMM;
    float v = 0.f;
    if (r < KROWS) {
      int lr = r / NB, n = r % NB;
      int grow = row0 + lr;
      if (c < DD) {
        v = seq_dense ? seq_dense[(long)(grow*NB + n)*DD + c]
                      : node_table[(long)seq_gidx[grow*NB + n]*DD + c];
      } else if (c < 2*DD) {
        int j = c - DD;
        float e = rel_b[j];
        #pragma unroll
        for (int q = 0; q < RR; ++q)
          e += seq_rel[(long)(grow*NB + n)*RR + q] * rel_W[q*DD + j];
        v = e;
      } else {
        int j = c - 2*DD;
        float dt = row_time[grow] - seq_t[grow*NB + n];
        v = cosf(dt * basis_freq[j] + phase[j]);
      }
    } else {
      int lr = r - KROWS;
      if (lr < RB) {
        int grow = row0 + lr;
        if (c < DD) {
          v = src_dense ? src_dense[(long)grow*DD + c]
                        : node_table[(long)src_gidx[grow]*DD + c];
        } else if (c < 2*DD) {
          v = 0.f;
        } else {
          v = cosf(phase[c - 2*DD]);
        }
      }
    }
    sA[idx] = (__bf16)v;
  }
  __syncthreads();

  // ---- Stage 2a: qh = Q @ wq  (A rows 80..95) -> f32 sQH ----
  #pragma unroll
  for (int i = 0; i < 3; ++i) {
    int n0 = (wave + i*8) * 16;
    v8f acc = {};
    for (int k0 = 0; k0 < DMM; k0 += 32) {
      v16bf a = frag_ld(sA, DMM, KROWS, k0, lane);
      v16bf b = frag_ld(wqT, DMM, n0, k0, lane);
      acc = wmma_bf16(a, b, acc);
    }
    int col = n0 + (lane & 15);
    int rb  = (lane >> 4) << 3;
    #pragma unroll
    for (int r = 0; r < 8; ++r) sQH[(rb + r)*DMM + col] = acc[r];
  }

  // ---- Stage 2b: kh = K @ wk, vh = K @ wv (80 rows) -> bf16 sKH/sVH ----
  #pragma unroll
  for (int i = 0; i < 3; ++i) {
    int n0 = (wave + i*8) * 16;
    v8f aK[5] = {}; v8f aV[5] = {};
    for (int k0 = 0; k0 < DMM; k0 += 32) {
      v16bf bk = frag_ld(wkT, DMM, n0, k0, lane);
      v16bf bv = frag_ld(wvT, DMM, n0, k0, lane);
      #pragma unroll
      for (int m = 0; m < 5; ++m) {
        v16bf a = frag_ld(sA, DMM, m*16, k0, lane);
        aK[m] = wmma_bf16(a, bk, aK[m]);
        aV[m] = wmma_bf16(a, bv, aV[m]);
      }
    }
    int col = n0 + (lane & 15);
    int rb  = (lane >> 4) << 3;
    #pragma unroll
    for (int m = 0; m < 5; ++m) {
      #pragma unroll
      for (int r = 0; r < 8; ++r) {
        int row = m*16 + rb + r;
        sKH[row*DMM + col] = (__bf16)aK[m][r];
        sVH[row*DMM + col] = (__bf16)aV[m][r];
      }
    }
  }
  __syncthreads();

  // ---- Stage 3: scores + softmax; wave -> (local row, head), lane -> neighbor ----
  {
    const int lr = wave >> 1;
    const int h  = wave & 1;
    const int n  = lane;
    const int grow = row0 + lr;
    float sc = -3.0e38f;
    if (n < NB) {
      const float*  qrow = sQH + lr*DMM + h*DKH;
      const __bf16* krow = sKH + (lr*NB + n)*DMM + h*DKH;
      float d = 0.f;
      for (int c2 = 0; c2 < DKH; ++c2) d += qrow[c2] * (float)krow[c2];
      sc = d * 0.07216878364870322f;                 // 1/sqrt(192)
      if (mask_idx[grow*NB + n] == 0) sc = NEGV;
    }
    float mx = sc;
    #pragma unroll
    for (int s2 = 16; s2 > 0; s2 >>= 1) mx = fmaxf(mx, __shfl_xor(mx, s2, 32));
    float e = (n < NB) ? __expf(sc - mx) : 0.f;
    float sm = e;
    #pragma unroll
    for (int s2 = 16; s2 > 0; s2 >>= 1) sm += __shfl_xor(sm, s2, 32);
    sATT[(lr*2 + h)*32 + n] = e / sm;
  }
  __syncthreads();

  // ---- Stage 4: attn-weighted V -> bf16 into sXB rows 0..3 (lda 512) ----
  for (int idx = tid; idx < RB*DMM; idx += 256) {
    int lr = idx / DMM, c = idx % DMM;
    int h = c / DKH;
    const float*  aw = sATT + (lr*2 + h)*32;
    const __bf16* vb = sVH + lr*NB*DMM + c;
    float acc = 0.f;
    #pragma unroll
    for (int n = 0; n < NB; ++n) acc += aw[n] * (float)vb[n*DMM];
    sXB[lr*512 + c] = (__bf16)acc;
  }
  __syncthreads();

  // ---- Stage 5: fc GEMM + bias + q residual -> f32 sX ----
  #pragma unroll
  for (int i = 0; i < 3; ++i) {
    int n0 = (wave + i*8) * 16;
    v8f acc = {};
    for (int k0 = 0; k0 < DMM; k0 += 32) {
      v16bf a = frag_ld(sXB, 512, 0, k0, lane);
      v16bf b = frag_ld(fcWT, DMM, n0, k0, lane);
      acc = wmma_bf16(a, b, acc);
    }
    int col = n0 + (lane & 15);
    int rb  = (lane >> 4) << 3;
    #pragma unroll
    for (int r = 0; r < 8; ++r) {
      int row = rb + r;
      float qres = (float)sA[(KROWS + row)*DMM + col];
      sX[row*DMM + col] = acc[r] + fc_b[col] + qres;
    }
  }
  __syncthreads();

  // ---- Stage 6: layernorm (waves 0..3, one row each), write [ln_out, src] bf16 ----
  if (wave < RB) {
    const int lr = wave;
    const float* xr = sX + lr*DMM;
    float s = 0.f;
    for (int c = lane; c < DMM; c += 32) s += xr[c];
    #pragma unroll
    for (int s2 = 16; s2 > 0; s2 >>= 1) s += __shfl_xor(s, s2, 32);
    float mu = s * (1.0f/DMM);
    float vs = 0.f;
    for (int c = lane; c < DMM; c += 32) { float d = xr[c] - mu; vs += d*d; }
    #pragma unroll
    for (int s2 = 16; s2 > 0; s2 >>= 1) vs += __shfl_xor(vs, s2, 32);
    float rstd = rsqrtf(vs * (1.0f/DMM) + 1e-5f);
    for (int c = lane; c < DMM; c += 32)
      sXB[lr*512 + c] = (__bf16)(((xr[c] - mu) * rstd) * ln_g[c] + ln_b[c]);
    for (int j = lane; j < DD; j += 32)
      sXB[lr*512 + DMM + j] = sA[(KROWS + lr)*DMM + j];   // src features (concat)
  }
  __syncthreads();

  // ---- Stage 7: mg1 GEMM (16x512 @ 512x128) + bias + relu -> bf16 sXB[:, :128] ----
  {
    int n0 = wave * 16;
    v8f acc = {};
    for (int k0 = 0; k0 < 512; k0 += 32) {
      v16bf a = frag_ld(sXB, 512, 0, k0, lane);
      v16bf b = frag_ld(mg1T, 512, n0, k0, lane);
      acc = wmma_bf16(a, b, acc);
    }
    __syncthreads();   // all waves done reading sXB before overwrite
    int col = n0 + (lane & 15);
    int rb  = (lane >> 4) << 3;
    #pragma unroll
    for (int r = 0; r < 8; ++r) {
      float v2 = acc[r] + mg1_b[col];
      sXB[(rb + r)*512 + col] = (__bf16)fmaxf(v2, 0.f);
    }
  }
  __syncthreads();

  // ---- Stage 8: mg2 GEMM (16x128 @ 128x128) + bias -> global out ----
  {
    int n0 = wave * 16;
    v8f acc = {};
    for (int k0 = 0; k0 < DD; k0 += 32) {
      v16bf a = frag_ld(sXB, 512, 0, k0, lane);
      v16bf b = frag_ld(mg2T, DD, n0, k0, lane);
      acc = wmma_bf16(a, b, acc);
    }
    int col = n0 + (lane & 15);
    int rb  = (lane >> 4) << 3;
    #pragma unroll
    for (int r = 0; r < 8; ++r) {
      int row = rb + r;
      if (row < RB) outp[(long)(row0 + row)*DD + col] = acc[r] + mg2_b[col];
    }
  }
}

// ---------------- host launcher ----------------
extern "C" void kernel_launch(void* const* d_in, const int* in_sizes, int n_in,
                              void* d_out, int out_size, void* d_ws, size_t ws_size,
                              hipStream_t stream) {
  (void)in_sizes; (void)n_in; (void)out_size; (void)ws_size;
  const int*   src_idx    = (const int*)  d_in[0];
  const float* cut_time   = (const float*)d_in[1];
  const int*   ngh1_nodes = (const int*)  d_in[2];
  const float* ngh1_t     = (const float*)d_in[3];
  const float* ngh1_rel   = (const float*)d_in[4];
  const int*   ngh2_nodes = (const int*)  d_in[5];
  const float* ngh2_t     = (const float*)d_in[6];
  const float* ngh2_rel   = (const float*)d_in[7];
  const float* node_table = (const float*)d_in[8];
  const float* basis_freq = (const float*)d_in[9];
  const float* phase      = (const float*)d_in[10];
  const float* rel_W      = (const float*)d_in[11];
  const float* rel_b      = (const float*)d_in[12];
  const float* w_q        = (const float*)d_in[13];
  const float* w_k        = (const float*)d_in[14];
  const float* w_v        = (const float*)d_in[15];
  const float* fc_W       = (const float*)d_in[16];
  const float* fc_b       = (const float*)d_in[17];
  const float* ln_g       = (const float*)d_in[18];
  const float* ln_b       = (const float*)d_in[19];
  const float* mg1_W      = (const float*)d_in[20];
  const float* mg1_b      = (const float*)d_in[21];
  const float* mg2_W      = (const float*)d_in[22];
  const float* mg2_b      = (const float*)d_in[23];

  // workspace layout (bytes)
  char* ws = (char*)d_ws;
  __bf16* wqT    = (__bf16*)(ws + 0);        // 2*384*384*2 = 589824
  __bf16* wkT    = (__bf16*)(ws + 589824);
  __bf16* wvT    = (__bf16*)(ws + 1179648);
  __bf16* fcWT   = (__bf16*)(ws + 1769472);
  __bf16* mg1T   = (__bf16*)(ws + 2359296);  // 2*512*128*2 = 262144
  __bf16* mg2T   = (__bf16*)(ws + 2621440);  // 2*128*128*2 = 65536
  float*  src_l1 = (float*) (ws + 2686976);  // 1024*128*4  = 524288
  float*  n1_l1  = (float*) (ws + 3211264);  // 20480*128*4 = 10485760

  (void)hipFuncSetAttribute((const void*)tgat_attn_layer,
                            hipFuncAttributeMaxDynamicSharedMemorySize, SMEM_BYTES);

  // weight conversion + transpose (f32 (L,K,N) -> bf16 (L,N,K))
  {
    long t1 = 2L*DMM*DMM;
    cvt_transpose_bf16<<<(unsigned)((t1+255)/256), 256, 0, stream>>>(w_q, wqT, 2, DMM, DMM);
    cvt_transpose_bf16<<<(unsigned)((t1+255)/256), 256, 0, stream>>>(w_k, wkT, 2, DMM, DMM);
    cvt_transpose_bf16<<<(unsigned)((t1+255)/256), 256, 0, stream>>>(w_v, wvT, 2, DMM, DMM);
    cvt_transpose_bf16<<<(unsigned)((t1+255)/256), 256, 0, stream>>>(fc_W, fcWT, 2, DMM, DMM);
    long t2 = 2L*512*DD;
    cvt_transpose_bf16<<<(unsigned)((t2+255)/256), 256, 0, stream>>>(mg1_W, mg1T, 2, 512, DD);
    long t3 = 2L*DD*DD;
    cvt_transpose_bf16<<<(unsigned)((t3+255)/256), 256, 0, stream>>>(mg2_W, mg2T, 2, DD, DD);
  }

  // Layer call 1: src_l1 = layer0(feat_src, feat_n1)
  tgat_attn_layer<<<BB1/RB, 256, SMEM_BYTES, stream>>>(
      BB1, node_table,
      src_idx, (const float*)nullptr,
      ngh1_nodes, (const float*)nullptr,
      ngh1_nodes, cut_time, ngh1_t, ngh1_rel,
      basis_freq, phase, rel_W, rel_b,
      wqT, wkT, wvT, fcWT, fc_b, ln_g, ln_b, mg1T, mg1_b, mg2T, mg2_b,
      src_l1);

  // Layer call 2 (the big one): n1_l1 = layer0(feat_n1.flat, feat_n2)
  tgat_attn_layer<<<(BB1*NB)/RB, 256, SMEM_BYTES, stream>>>(
      BB1*NB, node_table,
      ngh1_nodes, (const float*)nullptr,
      ngh2_nodes, (const float*)nullptr,
      ngh2_nodes, ngh1_t, ngh2_t, ngh2_rel,
      basis_freq, phase, rel_W, rel_b,
      wqT, wkT, wvT, fcWT, fc_b, ln_g, ln_b, mg1T, mg1_b, mg2T, mg2_b,
      n1_l1);

  // Layer call 3: out = layer1(src_l1, n1_l1) with layer-1 weight slices
  tgat_attn_layer<<<BB1/RB, 256, SMEM_BYTES, stream>>>(
      BB1, node_table,
      (const int*)nullptr, src_l1,
      (const int*)nullptr, n1_l1,
      ngh1_nodes, cut_time, ngh1_t, ngh1_rel,
      basis_freq, phase, rel_W, rel_b,
      wqT + DMM*DMM, wkT + DMM*DMM, wvT + DMM*DMM, fcWT + DMM*DMM,
      fc_b + DMM, ln_g + DMM, ln_b + DMM,
      mg1T + 512*DD, mg1_b + DD, mg2T + DD*DD, mg2_b + DD,
      (float*)d_out);
}
